// SNAIL_42923903156536
// MI455X (gfx1250) — matrix operand order repaired and checked by reading, not compiled, not run on hardware
//
#include <hip/hip_runtime.h>
#include <math.h>

typedef __bf16 bf16;
typedef __attribute__((ext_vector_type(16))) __bf16        v16bf;
typedef __attribute__((ext_vector_type(8)))  float         v8f;
typedef __attribute__((ext_vector_type(4)))  unsigned int  v4u;

union F16x16 { v4u q[2]; v16bf v; };   // 32 B: one WMMA 16-bit fragment
union F16x8  { v4u q;    bf16 e[8]; }; // 16 B: 8 bf16

#define BDIM 16
#define TDIM 1024
#define DDIM 256
#define CHS  2304
#define MDIM (BDIM * TDIM)   // 16384

// ---------------------------------------------------------------------------
// x (B,T,D) f32 -> H[b][c][t] f32 + bf16 mirror
// ---------------------------------------------------------------------------
__global__ __launch_bounds__(256)
void snail_transpose_in(const float* __restrict__ x, float* __restrict__ H,
                        bf16* __restrict__ Hb) {
  int i = blockIdx.x * 256 + threadIdx.x;
  if (i >= BDIM * TDIM * DDIM) return;
  int c = i % DDIM;
  int t = (i / DDIM) % TDIM;
  int b = i / (DDIM * TDIM);
  float v = x[i];
  size_t idx = ((size_t)b * CHS + c) * TDIM + t;
  H[idx]  = v;
  Hb[idx] = (bf16)v;
}

__global__ __launch_bounds__(256)
void snail_f32_to_bf16(const float* __restrict__ src, bf16* __restrict__ dst, int n) {
  int i = blockIdx.x * 256 + threadIdx.x;
  if (i < n) dst[i] = (bf16)src[i];
}

// conv weight (f=64, cin, 2) -> bf16 [cin][64] for one tap
__global__ __launch_bounds__(256)
void snail_conv_w(const float* __restrict__ w, bf16* __restrict__ dst, int cin, int tap) {
  int i = blockIdx.x * 256 + threadIdx.x;
  if (i >= cin * 64) return;
  int c = i >> 6, f = i & 63;
  dst[i] = (bf16)w[((size_t)f * cin + c) * 2 + tap];
}

// ---------------------------------------------------------------------------
// WMMA GEMM: out[m][n] = bias[n] + sum_taps sum_k A[b][k][t - shift_tap] * W_tap[k][n]
// A = bf16 activations in (B, CHS, T) layout, m = b*T + t.
// Block: 256 thr = 8 waves, 64x64 tile, K-step 32. LDS tiles are fragment-major:
// As[m][32k], Bs[n][32k] so each WMMA fragment = two ds_load_b128.
// outMode: 0 = bf16 [M][ldd], 1 = f32 [M][ldd], 2 = bf16 (B, N, T) transposed.
// ---------------------------------------------------------------------------
__global__ __launch_bounds__(256)
void snail_gemm(const bf16* __restrict__ A, int cin,
                const bf16* __restrict__ W0, const bf16* __restrict__ W1,
                int shift0, const float* __restrict__ bias, int N,
                float* __restrict__ dstF, bf16* __restrict__ dstB,
                int ldd, int outMode) {
  __shared__ bf16 As[64 * 32];
  __shared__ bf16 Bs[64 * 32];

  const int tid  = threadIdx.x;
  const int lane = tid & 31;
  const int wave = tid >> 5;
  const int wm = wave & 3, wn = wave >> 2;
  const int m0 = blockIdx.x * 64;
  const int n0 = blockIdx.y * 64;
  const int b  = m0 / TDIM;
  const int t0 = m0 % TDIM;
  const int lm = lane & 15;
  const bool hi = lane >= 16;
  const int kbA = hi ? 8 : 0;    // ISA 16-bit A 16x32 layout: K base per half-wave
  const int kbB = hi ? 8 : 0;    // B layout (same as A): K base per half-wave

  v8f acc0 = {0, 0, 0, 0, 0, 0, 0, 0};
  v8f acc1 = {0, 0, 0, 0, 0, 0, 0, 0};

  const int lrow = tid & 63;          // m (or n) handled by this thread
  const int ksl  = (tid >> 6) << 3;   // k sub-slice {0,8,16,24}

  for (int tap = 0; tap < 2; ++tap) {
    const bf16* W = tap ? W1 : W0;
    if (!W) break;
    const int shift = tap ? 0 : shift0;

    // clamped (branchless) A source position for this thread's m
    const int tA  = t0 + lrow - shift;
    const int tAc = tA < 0 ? 0 : tA;
    const int nB  = lrow < N ? lrow : 0;

    for (int k0 = 0; k0 < cin; k0 += 32) {
      __syncthreads();
      {  // A tile: 8 lane-coalesced loads (k-strided per thread), one b128 store
        const bf16* src = A + ((size_t)b * CHS + k0 + ksl) * TDIM + tAc;
        F16x8 buf;
        #pragma unroll
        for (int i = 0; i < 8; ++i) buf.e[i] = src[(size_t)i * TDIM];
        if (tA < 0) buf.q = (v4u){0, 0, 0, 0};
        *(v4u*)&As[lrow * 32 + ksl] = buf.q;
      }
      {  // B tile
        const bf16* wsrc = W + (size_t)(k0 + ksl) * N + nB;
        F16x8 buf;
        #pragma unroll
        for (int i = 0; i < 8; ++i) buf.e[i] = wsrc[(size_t)i * N];
        if (lrow >= N) buf.q = (v4u){0, 0, 0, 0};
        *(v4u*)&Bs[lrow * 32 + ksl] = buf.q;
      }
      __syncthreads();

      F16x16 au;
      {
        const bf16* ap = &As[(wm * 16 + lm) * 32 + kbA];
        au.q[0] = *(const v4u*)ap;          // K = kbA .. kbA+7
        au.q[1] = *(const v4u*)(ap + 16);   // K = kbA+16 .. kbA+23
      }
      #pragma unroll
      for (int nt = 0; nt < 2; ++nt) {
        F16x16 bu;
        const bf16* bp = &Bs[(wn * 32 + nt * 16 + lm) * 32 + kbB];
        bu.q[0] = *(const v4u*)bp;          // K = kbB .. kbB+7
        bu.q[1] = *(const v4u*)(bp + 16);   // K = kbB+16 .. kbB+23
        if (nt == 0)
          acc0 = __builtin_amdgcn_wmma_f32_16x16x32_bf16(false, au.v, false, bu.v,
                                                         (short)0, acc0, false, false);
        else
          acc1 = __builtin_amdgcn_wmma_f32_16x16x32_bf16(false, au.v, false, bu.v,
                                                         (short)0, acc1, false, false);
      }
    }
  }

  #pragma unroll
  for (int nt = 0; nt < 2; ++nt) {
    const int gn = n0 + wn * 32 + nt * 16 + lm;
    if (gn >= N) continue;
    const float bvv = bias ? bias[gn] : 0.0f;
    #pragma unroll
    for (int r = 0; r < 8; ++r) {
      const int gm = m0 + wm * 16 + r + (hi ? 8 : 0);
      const float v = (nt ? acc1[r] : acc0[r]) + bvv;
      if (outMode == 1)      dstF[(size_t)gm * ldd + gn] = v;
      else if (outMode == 0) dstB[(size_t)gm * ldd + gn] = (bf16)v;
      else {
        const int bb = gm / TDIM, tt = gm % TDIM;
        dstB[((size_t)bb * N + gn) * TDIM + tt] = (bf16)v;  // (B,N,T) for V
      }
    }
  }
}

// ---------------------------------------------------------------------------
// Causal flash attention. Q/K bf16 [B][T][ks]; V bf16 transposed [B][ks][T].
// grid = (T/128, B, ks/128). 8 independent waves, 16 query rows each,
// key blocks of 32, V features chunked by 128 (grid.z).
// Output scattered into H/Hb at channel offset coff, (B,C,T) layout.
// ---------------------------------------------------------------------------
__global__ __launch_bounds__(256)
void snail_flash(const bf16* __restrict__ Q, const bf16* __restrict__ K,
                 const bf16* __restrict__ Vt, int ks, float scale,
                 float* __restrict__ H, bf16* __restrict__ Hb, int coff) {
  __shared__ bf16 Ps[8][16 * 32];

  const int tid  = threadIdx.x;
  const int lane = tid & 31;
  const int wave = tid >> 5;
  const int lm = lane & 15;
  const bool hi = lane >= 16;
  const int kbA = hi ? 8 : 0;
  const int kbB = hi ? 8 : 0;
  const int b   = blockIdx.y;
  const int q0  = blockIdx.x * 128 + wave * 16;
  const int vc0 = blockIdx.z * 128;
  const size_t bT = (size_t)b * TDIM;
  const float NEGINF = -__builtin_inff();

  float rm[8], rl[8];
  v8f O[8];
  #pragma unroll
  for (int r = 0; r < 8; ++r) { rm[r] = NEGINF; rl[r] = 0.0f; }
  #pragma unroll
  for (int v = 0; v < 8; ++v) O[v] = (v8f){0, 0, 0, 0, 0, 0, 0, 0};

  bf16* myP = &Ps[wave][0];
  const bf16* qbase = Q + (bT + q0 + lm) * ks + kbA;

  for (int key0 = 0; key0 <= q0 + 15; key0 += 32) {
    // prefetch next key block (global_prefetch_b8)
    if (key0 + 32 <= q0 + 15) {
      __builtin_prefetch(K + (bT + key0 + 32 + lm) * ks, 0, 0);
      __builtin_prefetch(K + (bT + key0 + 48 + lm) * ks, 0, 0);
    }

    v8f s0 = {0, 0, 0, 0, 0, 0, 0, 0};
    v8f s1 = {0, 0, 0, 0, 0, 0, 0, 0};

    // S = Q K^T over feature chunks of 32 (all loads 16B-aligned vectors)
    for (int kc = 0; kc < ks; kc += 32) {
      F16x16 au;
      au.q[0] = *(const v4u*)(qbase + kc);
      au.q[1] = *(const v4u*)(qbase + kc + 16);
      #pragma unroll
      for (int nt = 0; nt < 2; ++nt) {
        int krow = key0 + nt * 16 + lm;
        if (krow > TDIM - 1) krow = TDIM - 1;   // masked anyway; keep in-bounds
        const bf16* kp = K + (bT + krow) * ks + kc + kbB;
        F16x16 bu;
        bu.q[0] = *(const v4u*)kp;
        bu.q[1] = *(const v4u*)(kp + 16);
        if (nt == 0)
          s0 = __builtin_amdgcn_wmma_f32_16x16x32_bf16(false, au.v, false, bu.v,
                                                       (short)0, s0, false, false);
        else
          s1 = __builtin_amdgcn_wmma_f32_16x16x32_bf16(false, au.v, false, bu.v,
                                                       (short)0, s1, false, false);
      }
    }

    // scale + causal mask + online softmax (half-wave shuffles; rows split lo/hi)
    #pragma unroll
    for (int r = 0; r < 8; ++r) {
      const int qg  = q0 + r + (hi ? 8 : 0);
      const int kj0 = key0 + lm;
      float v0 = s0[r] * scale;
      float v1 = s1[r] * scale;
      if (kj0      > qg) v0 = NEGINF;
      if (kj0 + 16 > qg) v1 = NEGINF;
      float mx = fmaxf(v0, v1);
      mx = fmaxf(mx, __shfl_xor(mx, 1, 32));
      mx = fmaxf(mx, __shfl_xor(mx, 2, 32));
      mx = fmaxf(mx, __shfl_xor(mx, 4, 32));
      mx = fmaxf(mx, __shfl_xor(mx, 8, 32));
      const float nm    = fmaxf(rm[r], mx);
      const float alpha = __expf(rm[r] - nm);
      const float p0 = __expf(v0 - nm);
      const float p1 = __expf(v1 - nm);
      s0[r] = p0; s1[r] = p1;
      float rs = p0 + p1;
      rs += __shfl_xor(rs, 1, 32);
      rs += __shfl_xor(rs, 2, 32);
      rs += __shfl_xor(rs, 4, 32);
      rs += __shfl_xor(rs, 8, 32);
      rl[r] = rl[r] * alpha + rs;
      rm[r] = nm;
      #pragma unroll
      for (int v = 0; v < 8; ++v) O[v][r] *= alpha;
    }

    // P (C-layout) -> per-wave LDS -> A-fragment layout
    #pragma unroll
    for (int r = 0; r < 8; ++r) {
      const int row = r + (hi ? 8 : 0);
      myP[row * 32 + lm]      = (bf16)s0[r];
      myP[row * 32 + 16 + lm] = (bf16)s1[r];
    }
    asm volatile("s_wait_dscnt 0" ::: "memory");
    F16x16 pu;
    {
      const bf16* pp = myP + lm * 32 + kbA;
      pu.q[0] = *(const v4u*)pp;
      pu.q[1] = *(const v4u*)(pp + 16);
    }

    // O += P * V : V transposed (B,ks,T) => two 8-key contiguous runs per lane
    const bool vfast = (key0 + 31 < TDIM);
    #pragma unroll
    for (int vs = 0; vs < 8; ++vs) {
      const int n = vc0 + vs * 16 + lm;
      const bf16* vp = Vt + ((size_t)b * ks + n) * TDIM + key0 + kbB;
      F16x16 vu;
      if (vfast) {
        vu.q[0] = *(const v4u*)vp;
        vu.q[1] = *(const v4u*)(vp + 16);
      } else {
        #pragma unroll
        for (int j = 0; j < 16; ++j) {
          int vrow = key0 + kbB + j + (j >= 8 ? 8 : 0);
          if (vrow > TDIM - 1) vrow = TDIM - 1;
          vu.v[j] = Vt[((size_t)b * ks + n) * TDIM + vrow];
        }
      }
      O[vs] = __builtin_amdgcn_wmma_f32_16x16x32_bf16(false, pu.v, false, vu.v,
                                                      (short)0, O[vs], false, false);
    }
  }

  // normalize rows, scatter into H (B,C,T) at channel offset coff
  #pragma unroll
  for (int vs = 0; vs < 8; ++vs) {
    const int n = vc0 + vs * 16 + lm;
    #pragma unroll
    for (int r = 0; r < 8; ++r) {
      const int qg = q0 + r + (hi ? 8 : 0);
      const float val = O[vs][r] / rl[r];
      const size_t idx = ((size_t)b * CHS + coff + n) * TDIM + qg;
      H[idx]  = val;
      Hb[idx] = (bf16)val;
    }
  }
}

// ---------------------------------------------------------------------------
// BatchNorm batch stats: y [M][64] -> stats[f] = {mean, rsqrt(var+eps)}
// ---------------------------------------------------------------------------
__global__ __launch_bounds__(256)
void snail_bn_stats(const float* __restrict__ y, float* __restrict__ stats) {
  __shared__ float ssum[256];
  __shared__ float ssq[256];
  const int f = blockIdx.x;
  float sum = 0.0f, sq = 0.0f;
  for (int m = threadIdx.x; m < MDIM; m += 256) {
    float v = y[(size_t)m * 64 + f];
    sum += v; sq += v * v;
  }
  ssum[threadIdx.x] = sum; ssq[threadIdx.x] = sq;
  __syncthreads();
  for (int s = 128; s > 0; s >>= 1) {
    if (threadIdx.x < s) {
      ssum[threadIdx.x] += ssum[threadIdx.x + s];
      ssq[threadIdx.x]  += ssq[threadIdx.x + s];
    }
    __syncthreads();
  }
  if (threadIdx.x == 0) {
    float mean = ssum[0] / (float)MDIM;
    float var  = ssq[0] / (float)MDIM - mean * mean;
    stats[2 * f]     = mean;
    stats[2 * f + 1] = rsqrtf(var + 1e-5f);
  }
}

// tanh(bn(y1)) * sigmoid(bn(y2)) -> H/Hb channels [coff, coff+64)
__global__ __launch_bounds__(256)
void snail_dense_finish(const float* __restrict__ y1, const float* __restrict__ y2,
                        const float* __restrict__ st1, const float* __restrict__ st2,
                        const float* __restrict__ g1, const float* __restrict__ be1,
                        const float* __restrict__ g2, const float* __restrict__ be2,
                        float* __restrict__ H, bf16* __restrict__ Hb, int coff) {
  int i = blockIdx.x * 256 + threadIdx.x;
  if (i >= MDIM * 64) return;
  int f = i & 63, m = i >> 6;
  float a = tanhf(g1[f] * (y1[i] - st1[2*f]) * st1[2*f + 1] + be1[f]);
  float z = g2[f] * (y2[i] - st2[2*f]) * st2[2*f + 1] + be2[f];
  float s = 1.0f / (1.0f + __expf(-z));
  float v = a * s;
  int b = m / TDIM, t = m % TDIM;
  size_t idx = ((size_t)b * CHS + coff + f) * TDIM + t;
  H[idx]  = v;
  Hb[idx] = (bf16)v;
}

// log_softmax over 10 classes; logits [M][10] -> out (B,T,NC) f32
__global__ __launch_bounds__(256)
void snail_out_head(const float* __restrict__ logits, float* __restrict__ out) {
  int m = blockIdx.x * 256 + threadIdx.x;
  if (m >= MDIM) return;
  const float* row = logits + (size_t)m * 10;
  float mx = row[0];
  #pragma unroll
  for (int n = 1; n < 10; ++n) mx = fmaxf(mx, row[n]);
  float s = 0.0f;
  #pragma unroll
  for (int n = 0; n < 10; ++n) s += __expf(row[n] - mx);
  float lse = mx + logf(s);
  #pragma unroll
  for (int n = 0; n < 10; ++n) out[(size_t)m * 10 + n] = row[n] - lse;
}

// ---------------------------------------------------------------------------
extern "C" void kernel_launch(void* const* d_in, const int* in_sizes, int n_in,
                              void* d_out, int out_size, void* d_ws, size_t ws_size,
                              hipStream_t stream) {
  (void)in_sizes; (void)n_in; (void)out_size; (void)ws_size;

  char* ws = (char*)d_ws;
  size_t off = 0;
  auto alloc = [&](size_t bytes) -> char* {
    char* p = ws + off;
    off += (bytes + 255) & ~(size_t)255;
    return p;
  };

  float* H      = (float*)alloc((size_t)BDIM * CHS * TDIM * 4);
  bf16*  Hb     = (bf16*) alloc((size_t)BDIM * CHS * TDIM * 2);
  bf16*  Qb     = (bf16*) alloc((size_t)MDIM * 384 * 2);
  bf16*  Kb     = (bf16*) alloc((size_t)MDIM * 384 * 2);
  bf16*  Vb     = (bf16*) alloc((size_t)MDIM * 384 * 2);   // transposed (B,ks,T)
  bf16*  W0s    = (bf16*) alloc((size_t)2304 * 384 * 2);
  bf16*  W1s    = (bf16*) alloc((size_t)2304 * 384 * 2);
  float* y1     = (float*)alloc((size_t)MDIM * 64 * 4);
  float* y2     = (float*)alloc((size_t)MDIM * 64 * 4);
  float* st1    = (float*)alloc(64 * 2 * 4);
  float* st2    = (float*)alloc(64 * 2 * 4);
  float* logits = (float*)alloc((size_t)MDIM * 10 * 4);

  int idx = 0;
  auto in = [&](void) { return (const float*)d_in[idx++]; };

  auto cvt = [&](const float* src, bf16* dst, int n) {
    snail_f32_to_bf16<<<(n + 255) / 256, 256, 0, stream>>>(src, dst, n);
  };
  auto gemm = [&](int cin, const bf16* Wt0, const bf16* Wt1, int shift0,
                  const float* bias, int N, int outMode,
                  float* dF, bf16* dB, int ldd) {
    dim3 grid(MDIM / 64, (N + 63) / 64);
    snail_gemm<<<grid, 256, 0, stream>>>(Hb, cin, Wt0, Wt1, shift0, bias, N,
                                         dF, dB, ldd, outMode);
  };

  const float* x = in();
  snail_transpose_in<<<(BDIM * TDIM * DDIM) / 256, 256, 0, stream>>>(x, H, Hb);

  int cin = 256;

  auto doAttn = [&](int ks) {
    const float* Wq = in(); const float* bq = in();
    const float* Wk = in(); const float* bk = in();
    const float* Wv = in(); const float* bv = in();
    cvt(Wq, W0s, cin * ks);
    gemm(cin, W0s, nullptr, 0, bq, ks, 0, nullptr, Qb, ks);
    cvt(Wk, W0s, cin * ks);
    gemm(cin, W0s, nullptr, 0, bk, ks, 0, nullptr, Kb, ks);
    cvt(Wv, W0s, cin * ks);
    gemm(cin, W0s, nullptr, 0, bv, ks, 2, nullptr, Vb, ks);  // (B,ks,T)
    float scale = 1.0f / sqrtf((float)ks);
    dim3 g(TDIM / 128, BDIM, ks / 128);
    snail_flash<<<g, 256, 0, stream>>>(Qb, Kb, Vb, ks, scale, H, Hb, cin);
    cin += ks;
  };

  auto doDense = [&](int dil) {
    const float* w1 = in(); const float* b1 = in();
    const float* g1 = in(); const float* be1 = in();
    const float* w2 = in(); const float* b2 = in();
    const float* g2 = in(); const float* be2 = in();
    int nw = cin * 64, nb = (nw + 255) / 256;
    snail_conv_w<<<nb, 256, 0, stream>>>(w1, W0s, cin, 0);
    snail_conv_w<<<nb, 256, 0, stream>>>(w1, W1s, cin, 1);
    gemm(cin, W0s, W1s, dil, b1, 64, 1, y1, nullptr, 64);
    snail_conv_w<<<nb, 256, 0, stream>>>(w2, W0s, cin, 0);
    snail_conv_w<<<nb, 256, 0, stream>>>(w2, W1s, cin, 1);
    gemm(cin, W0s, W1s, dil, b2, 64, 1, y2, nullptr, 64);
    snail_bn_stats<<<64, 256, 0, stream>>>(y1, st1);
    snail_bn_stats<<<64, 256, 0, stream>>>(y2, st2);
    snail_dense_finish<<<(MDIM * 64) / 256, 256, 0, stream>>>(
        y1, y2, st1, st2, g1, be1, g2, be2, H, Hb, cin);
    cin += 64;
  };

  doAttn(128);                                    // -> cin 384
  for (int i = 0; i < 10; ++i) doDense(1 << i);   // -> cin 1024
  doAttn(256);                                    // -> cin 1280
  for (int i = 0; i < 10; ++i) doDense(1 << i);   // -> cin 1920
  doAttn(384);                                    // -> cin 2304

  const float* outW = in();
  const float* outb = in();
  cvt(outW, W0s, cin * 10);
  gemm(cin, W0s, nullptr, 0, outb, 10, 1, logits, nullptr, 10);
  snail_out_head<<<MDIM / 256, 256, 0, stream>>>(logits, (float*)d_out);
}
